// FlowLSTM_52664888983562
// MI455X (gfx1250) — matrix operand order repaired and checked by reading, not compiled, not run on hardware
//
#include <hip/hip_runtime.h>
#include <hip/hip_bf16.h>
#include <stdint.h>

// ---------------------------------------------------------------------------
// FlowLSTM on MI455X (gfx1250): per-step fused bf16-WMMA GEMM + LSTM update.
//   gates[B,4H] = [x_t | h] @ [W_ih | W_hh]^T   (K padded 529 -> 544)
//   accumulate f32 via v_wmma_f32_16x16x32_bf16, fuse sigmoid/tanh epilogue.
// Each wave: 32 M-rows x 128 N-cols -> each LDS B-fragment feeds 2 WMMAs
// (ds:wmma ratio 1:1, ~512B LDS per WMMA, sustainable on WGP LDS bandwidth).
// ---------------------------------------------------------------------------

typedef __attribute__((ext_vector_type(16))) __bf16 v16bf;
typedef __attribute__((ext_vector_type(8)))  float  v8f;

#define HDIM   512
#define IDIM   17
#define TSTEPS 19
#define BDIM   8192
#define KP     544            // 512 (h) + 32 (x padded)
#define LDSROWB 80            // 64B K-chunk row + 16B pad (conflict-free b128)

union FragBF {
  v16bf v;
  uint4 q[2];
};

__device__ __forceinline__ float fast_sigmoid(float x) {
  return 1.0f / (1.0f + __expf(-x));
}

// --------------------------- prep kernels ----------------------------------

__global__ void prep_weights(const float* __restrict__ W_ih,
                             const float* __restrict__ W_hh,
                             const float* __restrict__ b_ih,
                             const float* __restrict__ b_hh,
                             __bf16* __restrict__ Wc,
                             float* __restrict__ biasc) {
  int n = blockIdx.x;  // 0..2047 (gate-unit row)
  for (int k = threadIdx.x; k < KP; k += blockDim.x) {
    float v = 0.0f;
    if (k < HDIM)                 v = W_hh[(size_t)n * HDIM + k];
    else if ((k - HDIM) < IDIM)   v = W_ih[(size_t)n * IDIM + (k - HDIM)];
    Wc[(size_t)n * KP + k] = (__bf16)v;
  }
  if (threadIdx.x == 0) biasc[n] = b_ih[n] + b_hh[n];
}

__global__ void prep_x(const float* __restrict__ x, __bf16* __restrict__ xpad) {
  // xpad layout: [T][B][32] bf16, zero-padded past IDIM
  size_t idx = (size_t)blockIdx.x * blockDim.x + threadIdx.x;
  int k = (int)(idx & 31);
  size_t tb = idx >> 5;
  int b = (int)(tb % BDIM);
  int t = (int)(tb / BDIM);
  if (t >= TSTEPS) return;
  float v = (k < IDIM) ? x[((size_t)b * TSTEPS + t) * IDIM + k] : 0.0f;
  xpad[idx] = (__bf16)v;
}

__global__ void prep_state(const float* __restrict__ h0,
                           const float* __restrict__ c0,
                           __bf16* __restrict__ hbuf,
                           float* __restrict__ cbuf) {
  size_t i = (size_t)blockIdx.x * blockDim.x + threadIdx.x;  // B*H threads
  hbuf[i] = (__bf16)h0[i];
  cbuf[i] = c0[i];
}

// --------------------------- main step kernel ------------------------------
// Grid: (B/256, H/32). Block: 256 threads = 8 waves.
// Block tile: 256 batch rows x 32 hidden units x all 4 gates (128 scattered N).
// Wave w: rows [mBlock + 32w, +32): 2 A-frags x 8 N-frags = 16 accumulators.

__global__ __launch_bounds__(256)
void lstm_step(const __bf16* __restrict__ hPrev,   // [B,H] bf16
               const float*  __restrict__ cPrev,   // [B,H] f32
               const __bf16* __restrict__ xt,      // [B,32] bf16 (this step)
               const __bf16* __restrict__ Wc,      // [2048,KP] bf16
               const float*  __restrict__ biasc,   // [2048]
               __bf16* __restrict__ hNext,
               float*  __restrict__ cNext) {
  __shared__ unsigned char ldsW[128 * LDSROWB] __attribute__((aligned(16)));

  const int tid   = threadIdx.x;
  const int wave  = tid >> 5;
  const int lane  = tid & 31;
  const int lhalf = lane >> 4;      // K-half selector within fragments
  const int l16   = lane & 15;

  const int mBlock = blockIdx.x * 256;
  const int u0     = blockIdx.y * 32;       // hidden-unit base for this block
  const int mWave  = mBlock + wave * 32;

  // Cooperative W staging: thread -> (local col wr in [0,128), half wh)
  const int wr = tid >> 1;
  const int wh = tid & 1;
  const int gW = wr >> 5;                    // gate 0..3
  const int uW = wr & 31;                    // unit within chunk
  const char* wRowBase = (const char*)Wc + (size_t)(gW * HDIM + u0 + uW) * (KP * 2);
  unsigned char* ldsDst = ldsW + wr * LDSROWB + wh * 32;

  // A (activations) row bases: frag f covers rows mWave + f*16 + l16
  const char* aRowH0 = (const char*)hPrev + (size_t)(mWave + l16) * (HDIM * 2);
  const char* aRowH1 = aRowH0 + (size_t)16 * (HDIM * 2);
  const char* aRowX0 = (const char*)xt + (size_t)(mWave + l16) * 64;
  const char* aRowX1 = aRowX0 + (size_t)16 * 64;

  v8f acc[2][8];
  const v8f vzero = {0.f, 0.f, 0.f, 0.f, 0.f, 0.f, 0.f, 0.f};
#pragma unroll
  for (int f = 0; f < 2; ++f)
#pragma unroll
    for (int n = 0; n < 8; ++n) acc[f][n] = vzero;

  for (int kb = 0; kb < 17; ++kb) {
    __syncthreads();
    {
      const char* src = wRowBase + kb * 64 + wh * 32;
      uint4 w0 = *reinterpret_cast<const uint4*>(src);
      uint4 w1 = *reinterpret_cast<const uint4*>(src + 16);
      if (kb < 16) __builtin_prefetch(src + 64, 0, 1);  // global_prefetch next K-chunk
      *reinterpret_cast<uint4*>(ldsDst)      = w0;
      *reinterpret_cast<uint4*>(ldsDst + 16) = w1;
    }
    __syncthreads();

    // A fragments (16x32 bf16): lanes 0-15 hold K[0..7],K[16..23];
    // lanes 16-31 hold K[8..15],K[24..31]  (ISA 16-bit A layout)
    FragBF a0, a1;
    {
      const char* ar0 = (kb < 16) ? (aRowH0 + kb * 64) : aRowX0;
      const char* ar1 = (kb < 16) ? (aRowH1 + kb * 64) : aRowX1;
      a0.q[0] = *reinterpret_cast<const uint4*>(ar0 + lhalf * 16);
      a0.q[1] = *reinterpret_cast<const uint4*>(ar0 + 32 + lhalf * 16);
      a1.q[0] = *reinterpret_cast<const uint4*>(ar1 + lhalf * 16);
      a1.q[1] = *reinterpret_cast<const uint4*>(ar1 + 32 + lhalf * 16);
    }

#pragma unroll
    for (int n = 0; n < 8; ++n) {
      // B fragment (32x16 bf16): lane = N col, lanes 0-15 K[0..7],K[16..23]; 16-31 K[8..15],K[24..31]
      FragBF b;
      const unsigned char* br = ldsW + (size_t)(n * 16 + l16) * LDSROWB + lhalf * 16;
      b.q[0] = *reinterpret_cast<const uint4*>(br);
      b.q[1] = *reinterpret_cast<const uint4*>(br + 32);
      acc[0][n] = __builtin_amdgcn_wmma_f32_16x16x32_bf16(
          false, a0.v, false, b.v, (short)0, acc[0][n], false, false);
      acc[1][n] = __builtin_amdgcn_wmma_f32_16x16x32_bf16(
          false, a1.v, false, b.v, (short)0, acc[1][n], false, false);
    }
  }

  // Fused LSTM epilogue. C/D layout: VGPR r, lane l -> M = r + 8*(l>>4), N = l&15.
  // Frag (f,n) -> row m = mWave + f*16 + r + 8*lhalf,
  //               gate g = n>>1, unit u = u0 + (n&1)*16 + l16.
#pragma unroll
  for (int hn = 0; hn < 2; ++hn) {
    const int u = u0 + hn * 16 + l16;
    const float bi = biasc[0 * HDIM + u];
    const float bf = biasc[1 * HDIM + u];
    const float bg = biasc[2 * HDIM + u];
    const float bo = biasc[3 * HDIM + u];
#pragma unroll
    for (int f = 0; f < 2; ++f) {
#pragma unroll
      for (int r = 0; r < 8; ++r) {
        const int m = mWave + f * 16 + r + 8 * lhalf;
        const float gi = acc[f][0 + hn][r] + bi;
        const float gf = acc[f][2 + hn][r] + bf;
        const float gg = acc[f][4 + hn][r] + bg;
        const float go = acc[f][6 + hn][r] + bo;
        const size_t off = (size_t)m * HDIM + u;
        const float cOld = cPrev[off];
        const float cN = fast_sigmoid(gf) * cOld + fast_sigmoid(gi) * tanhf(gg);
        const float hN = fast_sigmoid(go) * tanhf(cN);
        cNext[off] = cN;
        hNext[off] = (__bf16)hN;
      }
    }
  }
}

// --------------------------- output projection -----------------------------
// out[b,t,:] = h_t[b,:] @ W_d^T + b_d   (N=17, K=512). One thread per batch row.

__global__ __launch_bounds__(256)
void proj_step(const __bf16* __restrict__ hCur,
               const float* __restrict__ Wd,    // [I,H] row-major
               const float* __restrict__ bd,    // [I]
               float* __restrict__ out,         // [B,T,I]
               int t) {
  __shared__ float ldsWd[IDIM * HDIM];
  for (int i = threadIdx.x; i < IDIM * HDIM; i += blockDim.x) ldsWd[i] = Wd[i];
  __syncthreads();

  const int b = blockIdx.x * blockDim.x + threadIdx.x;
  float accv[IDIM];
#pragma unroll
  for (int i = 0; i < IDIM; ++i) accv[i] = bd[i];

  const __bf16* hrow = hCur + (size_t)b * HDIM;
  for (int k = 0; k < HDIM; ++k) {
    const float hv = (float)hrow[k];
#pragma unroll
    for (int i = 0; i < IDIM; ++i) accv[i] += hv * ldsWd[i * HDIM + k];
  }
  float* orow = out + ((size_t)b * TSTEPS + t) * IDIM;
#pragma unroll
  for (int i = 0; i < IDIM; ++i) orow[i] = accv[i];
}

// --------------------------- launcher --------------------------------------

static inline size_t align256(size_t x) { return (x + 255) & ~(size_t)255; }

extern "C" void kernel_launch(void* const* d_in, const int* in_sizes, int n_in,
                              void* d_out, int out_size, void* d_ws, size_t ws_size,
                              hipStream_t stream) {
  (void)in_sizes; (void)n_in; (void)out_size; (void)ws_size;

  const float* x    = (const float*)d_in[0];
  const float* h0   = (const float*)d_in[1];
  const float* c0   = (const float*)d_in[2];
  const float* W_ih = (const float*)d_in[3];
  const float* W_hh = (const float*)d_in[4];
  const float* b_ih = (const float*)d_in[5];
  const float* b_hh = (const float*)d_in[6];
  const float* W_d  = (const float*)d_in[7];
  const float* b_d  = (const float*)d_in[8];
  float* out = (float*)d_out;

  // Workspace carve-up (~60 MB total)
  char* ws = (char*)d_ws;
  __bf16* Wc    = (__bf16*)ws; ws += align256((size_t)4 * HDIM * KP * 2);
  float*  biasc = (float*)ws;  ws += align256((size_t)4 * HDIM * 4);
  __bf16* xpad  = (__bf16*)ws; ws += align256((size_t)TSTEPS * BDIM * 32 * 2);
  __bf16* hA    = (__bf16*)ws; ws += align256((size_t)BDIM * HDIM * 2);
  __bf16* hB    = (__bf16*)ws; ws += align256((size_t)BDIM * HDIM * 2);
  float*  cA    = (float*)ws;  ws += align256((size_t)BDIM * HDIM * 4);
  float*  cB    = (float*)ws;  ws += align256((size_t)BDIM * HDIM * 4);

  // One-time packing/conversion
  prep_weights<<<4 * HDIM, 128, 0, stream>>>(W_ih, W_hh, b_ih, b_hh, Wc, biasc);
  {
    size_t total = (size_t)TSTEPS * BDIM * 32;
    prep_x<<<(unsigned)((total + 255) / 256), 256, 0, stream>>>(x, xpad);
  }
  prep_state<<<(BDIM * HDIM) / 256, 256, 0, stream>>>(h0, c0, hA, cA);

  // Rolled time loop: ping-pong state buffers
  __bf16* hbuf[2] = {hA, hB};
  float*  cbuf[2] = {cA, cB};
  int cur = 0;
  for (int t = 0; t < TSTEPS; ++t) {
    lstm_step<<<dim3(BDIM / 256, HDIM / 32), 256, 0, stream>>>(
        hbuf[cur], cbuf[cur], xpad + (size_t)t * BDIM * 32,
        Wc, biasc, hbuf[1 - cur], cbuf[1 - cur]);
    proj_step<<<BDIM / 256, 256, 0, stream>>>(hbuf[1 - cur], W_d, b_d, out, t);
    cur ^= 1;
  }
}
